// UserSubredditSAGE_28355374088833
// MI455X (gfx1250) — matrix operand order repaired and checked by reading, not compiled, not run on hardware
//
#include <hip/hip_runtime.h>
#include <hip/hip_bf16.h>
#include <cstdint>
#include <cstddef>

#define N_SUB   50000
#define N_USERS 200000
#define N_EDGES 1000000
#define D_IN    3072
#define HDIM    512

typedef __attribute__((ext_vector_type(16))) __bf16 v16bf;
typedef __attribute__((ext_vector_type(8)))  float  v8f;

#if __has_builtin(__builtin_amdgcn_tensor_load_to_lds) && __has_builtin(__builtin_amdgcn_s_wait_tensorcnt)
#define USE_TDM 1
typedef __attribute__((ext_vector_type(4))) unsigned int u32x4;
typedef __attribute__((ext_vector_type(8))) int i32x8;
typedef __attribute__((ext_vector_type(4))) int i32x4;
#else
#define USE_TDM 0
#endif

__device__ __forceinline__ unsigned short f2bf(float f) {
    unsigned u = __float_as_uint(f);
    u += 0x7FFFu + ((u >> 16) & 1u);   // round-to-nearest-even
    return (unsigned short)(u >> 16);
}

__device__ __forceinline__ unsigned pack2bf(float a, float b) {
#if __has_builtin(__builtin_amdgcn_cvt_pk_bf16_f32)
    typedef __attribute__((ext_vector_type(2))) __bf16 v2bf;
    v2bf r = __builtin_amdgcn_cvt_pk_bf16_f32(a, b);
    return __builtin_bit_cast(unsigned, r);
#elif __has_builtin(__builtin_amdgcn_perm)
    unsigned ua = __float_as_uint(a); ua += 0x7FFFu + ((ua >> 16) & 1u);
    unsigned ub = __float_as_uint(b); ub += 0x7FFFu + ((ub >> 16) & 1u);
    // result bytes: [ua.b2, ua.b3, ub.b2, ub.b3] -> one v_perm_b32
    return __builtin_amdgcn_perm(ub, ua, 0x07060302u);
#else
    return (unsigned)f2bf(a) | ((unsigned)f2bf(b) << 16);
#endif
}

union FragBF { v16bf v; uint4 q[2]; };

enum { EPI_F32 = 0, EPI_RELU_BF16 = 1 };

// LDS layout: 4 tile buffers of 128 rows x 40 bf16 elems (A0, A1, B0, B1)
#define TILE_ELEMS (128 * 40)
#define LA0 0
#define LA1 (TILE_ELEMS)
#define LB0 (2 * TILE_ELEMS)
#define LB1 (3 * TILE_ELEMS)

#if USE_TDM
// Async-copy a 128-row x 32-col bf16 tile (row stride `stride0` elems) into LDS
// at byte offset ldsOff, HW-padding each 64B row with 16B -> 40-elem LDS stride.
__device__ __forceinline__ void tdm_load_tile_bf16(
    const unsigned short* gsrc, unsigned ldsOff,
    unsigned dim0, unsigned dim1, unsigned stride0)
{
    unsigned long long ga = (unsigned long long)(uintptr_t)gsrc;
    u32x4 g0;
    g0[0] = 1u;                                            // count=1 (valid descriptor)
    g0[1] = ldsOff;                                        // lds_addr
    g0[2] = (unsigned)(ga & 0xFFFFFFFFull);                // global_addr[31:0]
    g0[3] = (unsigned)((ga >> 32) & 0x1FFFFFFull)          // global_addr[56:32]
          | (2u << 30);                                    // type=2 ("image")
    i32x8 g1;
    g1[0] = (1 << 16)      // data_size = 2B
          | (1 << 20)      // pad_enable
          | (3 << 22)      // pad_interval: 16 DWORDs (64B = one 32-elem row)
          | (3 << 25);     // pad_amount: 4 DWORDs (16B = 8 elems) -> 40-elem stride
    g1[1] = (int)((dim0 & 0xFFFFu) << 16);                               // tensor_dim0[15:0]
    g1[2] = (int)(((dim0 >> 16) & 0xFFFFu) | ((dim1 & 0xFFFFu) << 16));  // dim0[31:16], dim1[15:0]
    g1[3] = (int)(((dim1 >> 16) & 0xFFFFu) | (32u << 16));               // dim1[31:16], tile_dim0=32
    g1[4] = 128;                                                         // tile_dim1=128, tile_dim2=0
    g1[5] = (int)stride0;                                                // tensor_dim0_stride[31:0]
    g1[6] = 0;
    g1[7] = 0;
    i32x4 z4 = {0, 0, 0, 0};
    i32x8 z8 = {0, 0, 0, 0, 0, 0, 0, 0};
    __builtin_amdgcn_tensor_load_to_lds(g0, g1, z4, z4, z8, 0);
}
#endif

// ---------------- WMMA GEMM ----------------
// C[M x 512] = A0[M x K0] @ Bt0^T (+ A1[M x K1] @ Bt1^T) + bias
// Bt matrices are pre-transposed bf16 weights: Bt[n*K + k].
// Block tile 128x128, BK=32, 256 threads = 8 wave32s, wave tile 32x64 (2x4 WMMA).
// Double-buffered LDS (selected by integer offsets into ONE shared array so the
// compiler keeps addrspace(3) -> ds_load_b128); B (and bf16 A) tiles via TDM,
// f32 A via pipelined cvt staging. Rows padded to 40 elems (80B) for aligned b128.

template<typename TA>
__device__ __forceinline__ void gemm_phase(const TA* __restrict__ A,
    const unsigned short* __restrict__ Bt,
    int M, int lda, int K, int rowBase, int colBase,
    unsigned short* lds, v8f acc[2][4])
{
    const int tid  = threadIdx.x;
    const int lane = tid & 31;
    const int wave = tid >> 5;
    const int wm = wave & 3;       // M-group (4 x 32 rows)
    const int wn = wave >> 2;      // N-group (2 x 64 cols)
    const int lr = lane & 15;
    const int g  = lane >> 4;      // lane-half selects K half: k = 16*g + s (contiguous order, same for A and B)

    int aOff[2], bOff[4];
#pragma unroll
    for (int i = 0; i < 2; i++) aOff[i] = (wm * 32 + i * 16 + lr) * 40 + g * 16;
#pragma unroll
    for (int j = 0; j < 4; j++) bOff[j] = (wn * 64 + j * 16 + lr) * 40 + g * 16;

#if USE_TDM
    constexpr int TOPS = (sizeof(TA) == 4) ? 1 : 2;   // TDM ops per stage
    const unsigned ldsBase = (unsigned)(uintptr_t)lds;  // low 32 bits of generic ptr = LDS byte offset
    const int nIter = K >> 5;

    float4 rA[2][2];
    auto gloadA = [&](int tile) {
#pragma unroll
        for (int itc = 0; itc < 2; itc++) {
            int idx = tid + itc * 256;
            int row = idx >> 2, q = idx & 3;
            int grow = rowBase + row;
            if (grow < M) {
                const float* p = (const float*)A + (size_t)grow * lda + tile * 32 + q * 8;
                rA[itc][0] = *(const float4*)p;
                rA[itc][1] = *(const float4*)(p + 4);
                if ((tile + 1) * 32 < K) __builtin_prefetch((const void*)(p + 32), 0, 3);
            } else {
                rA[itc][0] = make_float4(0.f, 0.f, 0.f, 0.f);
                rA[itc][1] = make_float4(0.f, 0.f, 0.f, 0.f);
            }
        }
    };
    auto storeA = [&](int bufOff) {   // element offset of destination A buffer
#pragma unroll
        for (int itc = 0; itc < 2; itc++) {
            int idx = tid + itc * 256;
            int row = idx >> 2, q = idx & 3;
            uint4 u;
            u.x = pack2bf(rA[itc][0].x, rA[itc][0].y);
            u.y = pack2bf(rA[itc][0].z, rA[itc][0].w);
            u.z = pack2bf(rA[itc][1].x, rA[itc][1].y);
            u.w = pack2bf(rA[itc][1].z, rA[itc][1].w);
            *(uint4*)(lds + bufOff + row * 40 + q * 8) = u;
        }
    };
    auto issueTDM = [&](int tile, int buf) {
        // B tile: rows colBase..+127 of Bt (512 x K), cols tile*32..+31
        tdm_load_tile_bf16(Bt + (size_t)colBase * K + tile * 32,
                           ldsBase + 2u * (buf ? LB1 : LB0),
                           (unsigned)K, 512u, (unsigned)K);
        if constexpr (sizeof(TA) == 2) {
            // A tile (bf16 operand; buffer row-padded so full 128-row tile is in-bounds)
            tdm_load_tile_bf16((const unsigned short*)A + (size_t)rowBase * lda + tile * 32,
                               ldsBase + 2u * (buf ? LA1 : LA0),
                               (unsigned)lda, (unsigned)M, (unsigned)lda);
        }
    };

    // prologue: stage tile 0 into buffer 0
    if constexpr (sizeof(TA) == 4) gloadA(0);
    if (tid < 32) issueTDM(0, 0);
    if constexpr (sizeof(TA) == 4) storeA(LA0);

    for (int it = 0; it < nIter; ++it) {
        const int  b    = it & 1;
        const bool more = (it + 1 < nIter);
        if constexpr (sizeof(TA) == 4) { if (more) gloadA(it + 1); }
        if (tid < 32) {
            if (more) {
                issueTDM(it + 1, 1 - b);
                __builtin_amdgcn_s_wait_tensorcnt(TOPS);   // retire tile `it`'s TDM ops (in-order)
            } else {
                __builtin_amdgcn_s_wait_tensorcnt(0);
            }
        }
        __syncthreads();   // publish tile `it` (TDM + prior LDS stores)

        const int ao = b ? LA1 : LA0;
        const int bo = b ? LB1 : LB0;
        FragBF fa[2], fb[4];
#pragma unroll
        for (int i = 0; i < 2; i++) {
            fa[i].q[0] = *(const uint4*)(lds + ao + aOff[i]);        // K g*16..g*16+7
            fa[i].q[1] = *(const uint4*)(lds + ao + aOff[i] + 8);    // K g*16+8..
        }
#pragma unroll
        for (int j = 0; j < 4; j++) {
            fb[j].q[0] = *(const uint4*)(lds + bo + bOff[j]);        // K g*16..g*16+7
            fb[j].q[1] = *(const uint4*)(lds + bo + bOff[j] + 8);    // K g*16+8..
        }
#pragma unroll
        for (int i = 0; i < 2; i++)
#pragma unroll
            for (int j = 0; j < 4; j++)
                acc[i][j] = __builtin_amdgcn_wmma_f32_16x16x32_bf16(
                    false, fa[i].v, false, fb[j].v, (short)0, acc[i][j], false, false);

        __syncthreads();   // all waves done reading buffer b before it is restaged
        if constexpr (sizeof(TA) == 4) { if (more) storeA(b ? LA0 : LA1); }
    }
#else
    // -------- fallback: single-buffered staging (no TDM builtin) --------
    for (int k0 = 0; k0 < K; k0 += 32) {
        __syncthreads();
#pragma unroll
        for (int it = 0; it < 2; it++) {
            int idx = tid + it * 256;
            int row = idx >> 2, q = idx & 3;
            *(uint4*)(lds + LB0 + row * 40 + q * 8) =
                *(const uint4*)(Bt + (size_t)(colBase + row) * K + k0 + q * 8);
        }
#pragma unroll
        for (int it = 0; it < 2; it++) {
            int idx = tid + it * 256;
            int row = idx >> 2, q = idx & 3;
            int grow = rowBase + row;
            uint4 u = {0, 0, 0, 0};
            if (grow < M) {
                if constexpr (sizeof(TA) == 4) {
                    const float* p = (const float*)A + (size_t)grow * lda + k0 + q * 8;
                    const float4 f0 = *(const float4*)p;
                    const float4 f1 = *(const float4*)(p + 4);
                    u.x = pack2bf(f0.x, f0.y); u.y = pack2bf(f0.z, f0.w);
                    u.z = pack2bf(f1.x, f1.y); u.w = pack2bf(f1.z, f1.w);
                } else {
                    u = *(const uint4*)((const unsigned short*)A + (size_t)grow * lda + k0 + q * 8);
                }
            }
            *(uint4*)(lds + LA0 + row * 40 + q * 8) = u;
        }
        __syncthreads();

        FragBF fa[2], fb[4];
#pragma unroll
        for (int i = 0; i < 2; i++) {
            fa[i].q[0] = *(const uint4*)(lds + LA0 + aOff[i]);
            fa[i].q[1] = *(const uint4*)(lds + LA0 + aOff[i] + 8);
        }
#pragma unroll
        for (int j = 0; j < 4; j++) {
            fb[j].q[0] = *(const uint4*)(lds + LB0 + bOff[j]);
            fb[j].q[1] = *(const uint4*)(lds + LB0 + bOff[j] + 8);
        }
#pragma unroll
        for (int i = 0; i < 2; i++)
#pragma unroll
            for (int j = 0; j < 4; j++)
                acc[i][j] = __builtin_amdgcn_wmma_f32_16x16x32_bf16(
                    false, fa[i].v, false, fb[j].v, (short)0, acc[i][j], false, false);
    }
    __syncthreads();
#endif
}

template<typename TA0, typename TA1, bool DUAL, int EPI>
__global__ void __launch_bounds__(256) gemm_kernel(
    const void* A0_, const unsigned short* __restrict__ Bt0,
    const void* A1_, const unsigned short* __restrict__ Bt1,
    const float* __restrict__ bias, void* Out,
    int M, int lda0, int lda1, int K0, int K1)
{
    __shared__ __align__(16) unsigned short lds[4 * TILE_ELEMS];
    const int rowBase = blockIdx.x * 128;
    const int colBase = blockIdx.y * 128;

    v8f acc[2][4];
#pragma unroll
    for (int i = 0; i < 2; i++)
#pragma unroll
        for (int j = 0; j < 4; j++)
#pragma unroll
            for (int e = 0; e < 8; e++) acc[i][j][e] = 0.0f;

    gemm_phase<TA0>((const TA0*)A0_, Bt0, M, lda0, K0, rowBase, colBase, lds, acc);
    if constexpr (DUAL)
        gemm_phase<TA1>((const TA1*)A1_, Bt1, M, lda1, K1, rowBase, colBase, lds, acc);

    const int tid = threadIdx.x, lane = tid & 31, wave = tid >> 5;
    const int wm = wave & 3, wn = wave >> 2, lr = lane & 15, g = lane >> 4;
#pragma unroll
    for (int i = 0; i < 2; i++) {
#pragma unroll
        for (int j = 0; j < 4; j++) {
            const int col = colBase + wn * 64 + j * 16 + lr;
            const float bv = bias ? bias[col] : 0.0f;
#pragma unroll
            for (int v = 0; v < 8; v++) {
                const int row = rowBase + wm * 32 + i * 16 + g * 8 + v;  // C layout: VGPR v -> M=v (+8 hi half)
                if (row < M) {
                    float x = acc[i][j][v] + bv;
                    if constexpr (EPI == EPI_RELU_BF16) {
                        x = fmaxf(x, 0.0f);
                        ((unsigned short*)Out)[(size_t)row * HDIM + col] = f2bf(x);
                    } else {
                        ((float*)Out)[(size_t)row * HDIM + col] = x;
                    }
                }
            }
        }
    }
}

// ---------------- one-time weight convert + transpose (f32 [K x 512] -> bf16 [512 x K]) ----------------
__global__ void __launch_bounds__(256) wtrans_kernel(const float* __restrict__ in,
                                                     unsigned short* __restrict__ out, int K)
{
    __shared__ float tile[32][33];
    const int kb = blockIdx.x * 32, nb = blockIdx.y * 32;
    const int tx = threadIdx.x, ty = threadIdx.y;   // (32, 8)
#pragma unroll
    for (int i = 0; i < 32; i += 8)
        tile[ty + i][tx] = in[(size_t)(kb + ty + i) * HDIM + nb + tx];
    __syncthreads();
#pragma unroll
    for (int i = 0; i < 32; i += 8)
        out[(size_t)(nb + ty + i) * K + kb + tx] = f2bf(tile[tx][ty + i]);
}

// ---------------- CSR build + aggregation ----------------

__global__ void zero_i32(int* p, int n) {
    int i = blockIdx.x * blockDim.x + threadIdx.x;
    if (i < n) p[i] = 0;
}

__global__ void hist_kernel(const int* __restrict__ dst, int* __restrict__ cnt, int n) {
    int i = blockIdx.x * blockDim.x + threadIdx.x;
    if (i < n) atomicAdd(&cnt[dst[i]], 1);
}

__global__ void __launch_bounds__(1024) scan_kernel(const int* __restrict__ cnt,
                                                    int* __restrict__ rowPtr,
                                                    int* __restrict__ cursor, int n)
{
    __shared__ int buf[1024];
    __shared__ int carry;
    if (threadIdx.x == 0) carry = 0;
    __syncthreads();
    for (int base = 0; base < n; base += 1024) {
        int i = base + threadIdx.x;
        int v = (i < n) ? cnt[i] : 0;
        buf[threadIdx.x] = v;
        __syncthreads();
        for (int off = 1; off < 1024; off <<= 1) {
            int t = (threadIdx.x >= off) ? buf[threadIdx.x - off] : 0;
            __syncthreads();
            buf[threadIdx.x] += t;
            __syncthreads();
        }
        int excl = carry + buf[threadIdx.x] - v;
        if (i < n) { rowPtr[i] = excl; cursor[i] = excl; }
        __syncthreads();
        if (threadIdx.x == 1023) carry += buf[1023];
        __syncthreads();
    }
    if (threadIdx.x == 0) rowPtr[n] = carry;
}

__global__ void fill_kernel(const int* __restrict__ src, const int* __restrict__ dst,
                            int* __restrict__ cursor, int* __restrict__ srcSorted, int n)
{
    int i = blockIdx.x * blockDim.x + threadIdx.x;
    if (i < n) {
        int p = atomicAdd(&cursor[dst[i]], 1);
        srcSorted[p] = src[i];
    }
}

// one wave32 per destination row: lane covers 4 floats x 4 strided chunks (512 total)
__global__ void __launch_bounds__(256) agg_kernel(const float* __restrict__ sub,
    const int* __restrict__ srcSorted, const int* __restrict__ rowPtr,
    unsigned short* __restrict__ meanBf, int nDst)
{
    int wave = threadIdx.x >> 5, lane = threadIdx.x & 31;
    int d = blockIdx.x * 8 + wave;
    if (d >= nDst) return;
    int e0 = rowPtr[d], e1 = rowPtr[d + 1];
    float ax[4][4];
#pragma unroll
    for (int t = 0; t < 4; t++)
#pragma unroll
        for (int c = 0; c < 4; c++) ax[t][c] = 0.0f;

    for (int e = e0; e < e1; e++) {
        int s = srcSorted[e];
        const float4* p = (const float4*)(sub + (size_t)s * HDIM);
#pragma unroll
        for (int t = 0; t < 4; t++) {
            float4 v = p[t * 32 + lane];
            ax[t][0] += v.x; ax[t][1] += v.y; ax[t][2] += v.z; ax[t][3] += v.w;
        }
    }
    float inv = (e1 > e0) ? (1.0f / (float)(e1 - e0)) : 0.0f;
#pragma unroll
    for (int t = 0; t < 4; t++) {
        uint2 u;
        u.x = pack2bf(ax[t][0] * inv, ax[t][1] * inv);
        u.y = pack2bf(ax[t][2] * inv, ax[t][3] * inv);
        *(uint2*)(meanBf + (size_t)d * HDIM + (size_t)(t * 32 + lane) * 4) = u;
    }
}

// ---------------- row L2-normalize (in place, 512 cols) ----------------
__global__ void __launch_bounds__(128) rownorm_kernel(float* X, int rows)
{
    __shared__ float red[4];
    __shared__ float sInv;
    int row = blockIdx.x;
    if (row >= rows) return;
    int tid = threadIdx.x, lane = tid & 31, wave = tid >> 5;
    float4 v = *(float4*)(X + (size_t)row * HDIM + tid * 4);
    float ss = v.x * v.x + v.y * v.y + v.z * v.z + v.w * v.w;
    for (int o = 16; o > 0; o >>= 1) ss += __shfl_down(ss, o, 32);
    if (lane == 0) red[wave] = ss;
    __syncthreads();
    if (tid == 0) {
        float t = red[0] + red[1] + red[2] + red[3];
        sInv = 1.0f / fmaxf(sqrtf(t), 1e-12f);
    }
    __syncthreads();
    float inv = sInv;
    v.x *= inv; v.y *= inv; v.z *= inv; v.w *= inv;
    *(float4*)(X + (size_t)row * HDIM + tid * 4) = v;
}

extern "C" void kernel_launch(void* const* d_in, const int* in_sizes, int n_in,
                              void* d_out, int out_size, void* d_ws, size_t ws_size,
                              hipStream_t stream)
{
    (void)in_sizes; (void)n_in; (void)out_size; (void)ws_size;
    const float* x_sub  = (const float*)d_in[0];
    const float* x_user = (const float*)d_in[1];
    const int*   e_src  = (const int*)d_in[2];
    const int*   e_dst  = (const int*)d_in[3];
    // d_in[4] edge_attr, d_in[13..16] edge-MLP params: output discarded by reference -> skipped
    const float* W_proj = (const float*)d_in[5];
    const float* b_proj = (const float*)d_in[6];
    const float* W1l = (const float*)d_in[7];
    const float* b1l = (const float*)d_in[8];
    const float* W1r = (const float*)d_in[9];
    const float* W2l = (const float*)d_in[10];
    const float* b2l = (const float*)d_in[11];
    const float* W2r = (const float*)d_in[12];

    float* subOut  = (float*)d_out;                          // [N_SUB x H]
    float* userOut = (float*)d_out + (size_t)N_SUB * HDIM;   // [N_USERS x H]

    char* w = (char*)d_ws;
    auto alloc = [&](size_t bytes) -> char* {
        char* p = w; w += (bytes + 255) & ~(size_t)255; return p;
    };
    int* cnt       = (int*)alloc((size_t)N_USERS * 4);
    int* rowPtr    = (int*)alloc(((size_t)N_USERS + 1) * 4);
    int* cursor    = (int*)alloc((size_t)N_USERS * 4);
    int* srcSorted = (int*)alloc((size_t)N_EDGES * 4);
    // +128 rows padding so full 128-row TDM tiles stay inside the workspace
    unsigned short* meanBf = (unsigned short*)alloc((size_t)(N_USERS + 128) * HDIM * 2);
    unsigned short* u1Bf   = (unsigned short*)alloc((size_t)(N_USERS + 128) * HDIM * 2);
    unsigned short* WtProj = (unsigned short*)alloc((size_t)HDIM * D_IN * 2);
    unsigned short* Wt1l   = (unsigned short*)alloc((size_t)HDIM * HDIM * 2);
    unsigned short* Wt1r   = (unsigned short*)alloc((size_t)HDIM * HDIM * 2);
    unsigned short* Wt2l   = (unsigned short*)alloc((size_t)HDIM * HDIM * 2);
    unsigned short* Wt2r   = (unsigned short*)alloc((size_t)HDIM * HDIM * 2);

    // one-time weight convert+transpose: Wt[n][k] = bf16(W[k][n])
    {
        dim3 blk(32, 8);
        wtrans_kernel<<<dim3(D_IN / 32, HDIM / 32), blk, 0, stream>>>(W_proj, WtProj, D_IN);
        wtrans_kernel<<<dim3(HDIM / 32, HDIM / 32), blk, 0, stream>>>(W1l, Wt1l, HDIM);
        wtrans_kernel<<<dim3(HDIM / 32, HDIM / 32), blk, 0, stream>>>(W1r, Wt1r, HDIM);
        wtrans_kernel<<<dim3(HDIM / 32, HDIM / 32), blk, 0, stream>>>(W2l, Wt2l, HDIM);
        wtrans_kernel<<<dim3(HDIM / 32, HDIM / 32), blk, 0, stream>>>(W2r, Wt2r, HDIM);
    }

    // CSR build (independent of GEMM1; stream-ordered)
    zero_i32<<<(N_USERS + 255) / 256, 256, 0, stream>>>(cnt, N_USERS);
    hist_kernel<<<(N_EDGES + 255) / 256, 256, 0, stream>>>(e_dst, cnt, N_EDGES);
    scan_kernel<<<1, 1024, 0, stream>>>(cnt, rowPtr, cursor, N_USERS);
    fill_kernel<<<(N_EDGES + 255) / 256, 256, 0, stream>>>(e_src, e_dst, cursor, srcSorted, N_EDGES);

    // sub = normalize(x_sub @ W_proj + b_proj)
    {
        dim3 grid((N_SUB + 127) / 128, HDIM / 128);
        gemm_kernel<float, float, false, EPI_F32><<<grid, 256, 0, stream>>>(
            x_sub, WtProj, nullptr, nullptr, b_proj, subOut, N_SUB, D_IN, 0, D_IN, 0);
    }
    rownorm_kernel<<<N_SUB, 128, 0, stream>>>(subOut, N_SUB);

    // mean = segment_mean(sub[src]) over edges  (shared by both convs)
    agg_kernel<<<(N_USERS + 7) / 8, 256, 0, stream>>>(subOut, srcSorted, rowPtr, meanBf, N_USERS);

    // u1 = relu(mean @ W1l + x_user @ W1r + b1l)  -> bf16
    {
        dim3 grid((N_USERS + 127) / 128, HDIM / 128);
        gemm_kernel<unsigned short, float, true, EPI_RELU_BF16><<<grid, 256, 0, stream>>>(
            meanBf, Wt1l, x_user, Wt1r, b1l, u1Bf, N_USERS, HDIM, HDIM, HDIM, HDIM);
    }
    // user = normalize(mean @ W2l + u1 @ W2r + b2l)
    {
        dim3 grid((N_USERS + 127) / 128, HDIM / 128);
        gemm_kernel<unsigned short, unsigned short, true, EPI_F32><<<grid, 256, 0, stream>>>(
            meanBf, Wt2l, u1Bf, Wt2r, b2l, userOut, N_USERS, HDIM, HDIM, HDIM, HDIM);
    }
    rownorm_kernel<<<N_USERS, 128, 0, stream>>>(userOut, N_USERS);
}
